// MultiHeadAttention_86268713107886
// MI455X (gfx1250) — matrix operand order repaired and checked by reading, not compiled, not run on hardware
//
#include <hip/hip_runtime.h>

// Problem constants (match reference setup_inputs)
#define D_MODEL 1024
#define N_HEADS 16
#define D_HEAD  64
#define SEQ     2048
#define BATCH   2
#define M_TOTAL (BATCH * SEQ)   // 4096 rows for the projection GEMMs

typedef __attribute__((ext_vector_type(16))) __bf16 bf16x16;
typedef __attribute__((ext_vector_type(8)))  __bf16 bf16x8;
typedef __attribute__((ext_vector_type(8)))  float  f32x8;
typedef __attribute__((ext_vector_type(4)))  float  f32x4;
typedef __attribute__((ext_vector_type(4)))  unsigned int u32x4;
typedef __attribute__((ext_vector_type(8)))  int    i32x8;
typedef __attribute__((ext_vector_type(4)))  int    i32x4;

#if __has_builtin(__builtin_amdgcn_tensor_load_to_lds) && \
    __has_builtin(__builtin_amdgcn_s_wait_tensorcnt)
#define HAVE_TDM 1
#else
#define HAVE_TDM 0
#endif

static __device__ __forceinline__ f32x8 wmma_bf16(bf16x16 a, bf16x16 b, f32x8 c) {
  // D = A(16x32 bf16) * B(32x16 bf16) + C(16x16 f32)
  return __builtin_amdgcn_wmma_f32_16x16x32_bf16(false, a, false, b, (short)0, c,
                                                 false, false);
}

// ---------------------------------------------------------------------------
// Operand tile loaders (wave32). Work for both global and LDS pointers.
// A operand: 16(M) x 32(K), row-major source. Lane = M row (lane&15).
//   half 0: K[0..7] + K[16..23]; half 1: K[8..15] + K[24..31]
// B operand: 32(K) x 16(N), source organized N-major (16 rows of >=32 K elems).
//   Lane = N column (lane&15). half 0: K[0..7] + K[16..23]; half 1: K[8..15] + K[24..31]
// Strides in elements, multiples of 8 (16B alignment).
// ---------------------------------------------------------------------------
static __device__ __forceinline__ bf16x16 load_a_bf16(const __bf16* base, int stride) {
  const int lane = threadIdx.x & 31;
  const __bf16* p = base + (size_t)(lane & 15) * stride + ((lane >> 4) << 3);
  bf16x8 lo = *(const bf16x8*)p;
  bf16x8 hi = *(const bf16x8*)(p + 16);
  bf16x16 r;
#pragma unroll
  for (int i = 0; i < 8; ++i) { r[i] = lo[i]; r[i + 8] = hi[i]; }
  return r;
}

static __device__ __forceinline__ bf16x16 load_b_bf16(const __bf16* base, int stride) {
  const int lane = threadIdx.x & 31;
  const __bf16* p = base + (size_t)(lane & 15) * stride + ((lane >> 4) << 3);
  bf16x8 lo = *(const bf16x8*)p;
  bf16x8 hi = *(const bf16x8*)(p + 16);
  bf16x16 r;
#pragma unroll
  for (int i = 0; i < 8; ++i) { r[i] = lo[i]; r[i + 8] = hi[i]; }
  return r;
}

// ---------------------------------------------------------------------------
// fp32 -> bf16 bulk convert (8 elements per thread)
// ---------------------------------------------------------------------------
__global__ __launch_bounds__(256) void mha_cvt_bf16_kernel(
    const float* __restrict__ in, __bf16* __restrict__ out, int n8) {
  const int i = blockIdx.x * 256 + threadIdx.x;
  if (i < n8) {
    f32x4 a = *(const f32x4*)(in + (size_t)i * 8);
    f32x4 b = *(const f32x4*)(in + (size_t)i * 8 + 4);
    bf16x8 o;
#pragma unroll
    for (int k = 0; k < 4; ++k) { o[k] = (__bf16)a[k]; o[k + 4] = (__bf16)b[k]; }
    *(bf16x8*)(out + (size_t)i * 8) = o;
  }
}

// ---------------------------------------------------------------------------
// Tensor Data Mover: 2D tile (tile_d0 contiguous elems x tile_d1 rows,
// row stride = t_d0_stride elems, 2-byte elements) global -> LDS (packed).
// D# layout per CDNA5 ISA ch.8 (group0: count/lds/global/type, group1: dims).
// This toolchain exposes the 6-arg builtin: (g0 v4u32, g1 v8i32, v4i32, v4i32,
// v8i32, i32 cpol) — extra groups zero for <=2D tiles.
// ---------------------------------------------------------------------------
#if HAVE_TDM
static __device__ __forceinline__ void tdm_load_2d(
    unsigned lds_addr, const void* gptr,
    unsigned tile_d0, unsigned tile_d1, unsigned long long t_d0_stride) {
  const unsigned long long ga = (unsigned long long)(uintptr_t)gptr;
  const unsigned td0 = 0x40000000u;   // huge tensor dims: no OOB clipping
  const unsigned td1 = 0x40000000u;
  u32x4 g0;
  g0[0] = 1u;                                        // count=1, user mode
  g0[1] = lds_addr;                                  // LDS byte address
  g0[2] = (unsigned)(ga & 0xffffffffu);              // global_addr[31:0]
  g0[3] = (unsigned)((ga >> 32) & 0x1ffffffu) | (2u << 30);  // addr[56:32]|type=2
  i32x8 g1;
  g1[0] = (int)(1u << 16);                           // data_size=1 (2 bytes)
  g1[1] = (int)((td0 & 0xffffu) << 16);              // tensor_dim0[15:0] @ [63:48]
  g1[2] = (int)((td0 >> 16) | ((td1 & 0xffffu) << 16));  // td0[31:16], td1[15:0]
  g1[3] = (int)((td1 >> 16) | (tile_d0 << 16));      // td1[31:16], tile_dim0
  g1[4] = (int)(tile_d1 & 0xffffu);                  // tile_dim1 (tile_dim2=0)
  g1[5] = (int)(t_d0_stride & 0xffffffffu);          // dim0 stride[31:0]
  g1[6] = (int)((t_d0_stride >> 32) & 0xffffu);      // dim0 stride[47:32]
  g1[7] = 0;
  i32x4 z4 = {0, 0, 0, 0};
  i32x8 z8 = {0, 0, 0, 0, 0, 0, 0, 0};
  __builtin_amdgcn_tensor_load_to_lds(g0, g1, z4, z4, z8, 0);
}

static __device__ __forceinline__ void stage_chunk_tdm(
    __bf16* ldsK, __bf16* ldsV, const __bf16* Kg, const __bf16* Vg) {
  // K chunk: 32 key rows x 64 dh, contiguous in global -> LDS rows of 64
  tdm_load_2d((unsigned)(uintptr_t)ldsK, Kg, 64, 32, 64);
  // V chunk: 64 dh rows x 32 keys, global row stride SEQ -> LDS rows of 32
  tdm_load_2d((unsigned)(uintptr_t)ldsV, Vg, 32, 64, SEQ);
}
#else
static __device__ __forceinline__ void stage_chunk_copy(
    __bf16* ldsK, __bf16* ldsV, const __bf16* Kg, const __bf16* Vg, int tid) {
  for (int idx = tid; idx < 256; idx += 128)        // K: contiguous 4KB
    *(bf16x8*)(ldsK + idx * 8) = *(const bf16x8*)(Kg + idx * 8);
  for (int idx = tid; idx < 256; idx += 128) {      // V: 64 rows x 32 elems
    const int r = idx >> 2, c = (idx & 3) * 8;
    *(bf16x8*)(ldsV + r * 32 + c) = *(const bf16x8*)(Vg + (size_t)r * SEQ + c);
  }
}
#endif

// ---------------------------------------------------------------------------
// Projection GEMM (all-bf16 operands):  out = X @ W^T + bvec
// vmode 0: write bf16 [B,H,S,dh] (Q,K) ; vmode 1: write bf16 [B,H,dh,S] (V^T)
// Wave tile 16x128; block = 8 waves; grid (32, 8).
// ---------------------------------------------------------------------------
__global__ __launch_bounds__(256) void mha_proj_kernel(
    const __bf16* __restrict__ X, const __bf16* __restrict__ W,
    const float* __restrict__ bvec, __bf16* __restrict__ out, int vmode) {
  const int lane  = threadIdx.x & 31;
  const int wave  = threadIdx.x >> 5;
  const int half  = lane >> 4;
  const int nlane = lane & 15;
  const int m0 = blockIdx.x * 128 + wave * 16;
  const int n0 = blockIdx.y * 128;

  f32x8 C[8];
#pragma unroll
  for (int t = 0; t < 8; ++t)
#pragma unroll
    for (int r = 0; r < 8; ++r) C[t][r] = 0.f;

  for (int kk = 0; kk < D_MODEL; kk += 32) {
    bf16x16 a = load_a_bf16(X + (size_t)m0 * D_MODEL + kk, D_MODEL);
#pragma unroll
    for (int t = 0; t < 8; ++t) {
      bf16x16 bt = load_b_bf16(W + (size_t)(n0 + t * 16) * D_MODEL + kk, D_MODEL);
      C[t] = wmma_bf16(a, bt, C[t]);
    }
  }

#pragma unroll
  for (int t = 0; t < 8; ++t) {
    const int n = n0 + t * 16 + nlane;
    const float bb = bvec[n];
    const int h = n >> 6;
    const int d = n & (D_HEAD - 1);
#pragma unroll
    for (int r = 0; r < 8; ++r) {
      const int m = m0 + r + 8 * half;
      const int bi = m >> 11;
      const int s = m & (SEQ - 1);
      const float v = C[t][r] + bb;
      size_t off;
      if (vmode == 0)
        off = ((size_t)(bi * N_HEADS + h) * SEQ + s) * D_HEAD + d;
      else
        off = ((size_t)(bi * N_HEADS + h) * D_HEAD + d) * SEQ + s;
      out[off] = (__bf16)v;
    }
  }
}

// ---------------------------------------------------------------------------
// Flash attention. Block = 4 waves (64 query rows of one (b,h)); keys streamed
// in chunks of 32, staged in double-buffered LDS by the TDM (wave 0 issues,
// s_wait_tensorcnt + barriers synchronize). Grid (B*H=32, SEQ/64=32).
// ---------------------------------------------------------------------------
__global__ __launch_bounds__(128) void mha_flash_attn_kernel(
    const __bf16* __restrict__ Q, const __bf16* __restrict__ K,
    const __bf16* __restrict__ Vt, const float* __restrict__ bias,
    const int* __restrict__ mask, __bf16* __restrict__ AO) {
  __shared__ __bf16 ldsK[2][32 * 64];   // [key][dh] rows of 64, 4KB each
  __shared__ __bf16 ldsV[2][64 * 32];   // [dh][key] rows of 32, 4KB each
  __shared__ __bf16 plds[4][16 * 40];   // per-wave P transpose tile (80B rows)

  const int lane  = threadIdx.x & 31;
  const int wave  = threadIdx.x >> 5;
  const int half  = lane >> 4;
  const int nlane = lane & 15;
  const int bh = blockIdx.x;
  const int b  = bh >> 4;
  const int h  = bh & (N_HEADS - 1);
  const int q0 = blockIdx.y * 64 + wave * 16;

  const __bf16* qptr = Q + ((size_t)bh * SEQ + q0) * D_HEAD;
  bf16x16 qa0 = load_a_bf16(qptr, D_HEAD);        // dh 0..31
  bf16x16 qa1 = load_a_bf16(qptr + 32, D_HEAD);   // dh 32..63

  f32x8 O[4];
  f32x8 mrow, lrow;
#pragma unroll
  for (int r = 0; r < 8; ++r) { mrow[r] = -1e30f; lrow[r] = 0.f; }
#pragma unroll
  for (int t = 0; t < 4; ++t)
#pragma unroll
    for (int r = 0; r < 8; ++r) O[t][r] = 0.f;

  const float* biasB = bias + (size_t)b * SEQ * SEQ;
  const int*   maskB = mask + (size_t)b * SEQ * SEQ;
  const __bf16* Kbh = K + (size_t)bh * SEQ * D_HEAD;
  const __bf16* Vbh = Vt + (size_t)bh * D_HEAD * SEQ;
  __bf16* myLds = &plds[wave][0];

  // stage chunk 0
#if HAVE_TDM
  if (wave == 0) stage_chunk_tdm(&ldsK[0][0], &ldsV[0][0], Kbh, Vbh);
#else
  stage_chunk_copy(&ldsK[0][0], &ldsV[0][0], Kbh, Vbh, threadIdx.x);
  __syncthreads();
#endif

  for (int jc = 0; jc < SEQ / 32; ++jc) {
    const int j = jc * 32;
    const int cur = jc & 1;
    const int nxt = cur ^ 1;
    const bool hasNext = (j + 32) < SEQ;

#if HAVE_TDM
    if (wave == 0) {
      if (hasNext) {
        stage_chunk_tdm(&ldsK[nxt][0], &ldsV[nxt][0],
                        Kbh + (size_t)(j + 32) * D_HEAD, Vbh + j + 32);
        __builtin_amdgcn_s_wait_tensorcnt(2);   // current chunk's 2 DMAs done
      } else {
        __builtin_amdgcn_s_wait_tensorcnt(0);
      }
    }
    __syncthreads();   // release current chunk to all waves
#else
    if (hasNext)
      stage_chunk_copy(&ldsK[nxt][0], &ldsV[nxt][0],
                       Kbh + (size_t)(j + 32) * D_HEAD, Vbh + j + 32, threadIdx.x);
#endif

    const __bf16* kL = &ldsK[cur][0];
    bf16x16 kb00 = load_b_bf16(kL, 64);              // keys 0..15, dh 0..31
    bf16x16 kb01 = load_b_bf16(kL + 32, 64);         // keys 0..15, dh 32..63
    bf16x16 kb10 = load_b_bf16(kL + 16 * 64, 64);    // keys 16..31
    bf16x16 kb11 = load_b_bf16(kL + 16 * 64 + 32, 64);

    f32x8 s0, s1;
#pragma unroll
    for (int r = 0; r < 8; ++r) { s0[r] = 0.f; s1[r] = 0.f; }
    s0 = wmma_bf16(qa0, kb00, s0);
    s0 = wmma_bf16(qa1, kb01, s0);
    s1 = wmma_bf16(qa0, kb10, s1);
    s1 = wmma_bf16(qa1, kb11, s1);

    // scale (1/sqrt(64)) + chemical bias + mask
#pragma unroll
    for (int r = 0; r < 8; ++r) {
      const int m = q0 + r + 8 * half;
      const size_t rowoff = (size_t)m * SEQ + j + nlane;
      const float v0 = s0[r] * 0.125f + biasB[rowoff];
      const float v1 = s1[r] * 0.125f + biasB[rowoff + 16];
      s0[r] = maskB[rowoff]      ? v0 : -1e30f;
      s1[r] = maskB[rowoff + 16] ? v1 : -1e30f;
    }

    // online softmax (reductions stay inside 16-lane groups)
    f32x8 mnew, alpha;
#pragma unroll
    for (int r = 0; r < 8; ++r) {
      float cm = fmaxf(s0[r], s1[r]);
      cm = fmaxf(cm, __shfl_xor(cm, 1));
      cm = fmaxf(cm, __shfl_xor(cm, 2));
      cm = fmaxf(cm, __shfl_xor(cm, 4));
      cm = fmaxf(cm, __shfl_xor(cm, 8));
      mnew[r]  = fmaxf(mrow[r], cm);
      alpha[r] = __expf(mrow[r] - mnew[r]);
      mrow[r]  = mnew[r];
    }
#pragma unroll
    for (int r = 0; r < 8; ++r) {
      s0[r] = __expf(s0[r] - mnew[r]);
      s1[r] = __expf(s1[r] - mnew[r]);
      float rs = s0[r] + s1[r];
      rs += __shfl_xor(rs, 1);
      rs += __shfl_xor(rs, 2);
      rs += __shfl_xor(rs, 4);
      rs += __shfl_xor(rs, 8);
      lrow[r] = lrow[r] * alpha[r] + rs;
    }
#pragma unroll
    for (int t = 0; t < 4; ++t)
#pragma unroll
      for (int r = 0; r < 8; ++r) O[t][r] *= alpha[r];

    // P: C layout -> bf16 A layout via per-wave LDS tile (same-wave LDS in-order)
#pragma unroll
    for (int r = 0; r < 8; ++r) {
      const int row = r + 8 * half;
      myLds[row * 40 + nlane]      = (__bf16)s0[r];
      myLds[row * 40 + 16 + nlane] = (__bf16)s1[r];
    }
    asm volatile("" ::: "memory");
    bf16x16 pa = load_a_bf16(myLds, 40);
    asm volatile("" ::: "memory");

    // O += P(16x32) @ V(32x64) from staged V^T tile (rows of 32)
#pragma unroll
    for (int t = 0; t < 4; ++t) {
      bf16x16 vb = load_b_bf16(&ldsV[cur][0] + t * 16 * 32, 32);
      O[t] = wmma_bf16(pa, vb, O[t]);
    }

    __syncthreads();   // everyone done with current buffers before re-staging
  }

  // normalize and write attn output in [B, S, H*dh] row-major (bf16)
#pragma unroll
  for (int t = 0; t < 4; ++t) {
#pragma unroll
    for (int r = 0; r < 8; ++r) {
      const int m = q0 + r + 8 * half;
      const int d = t * 16 + nlane;
      const float v = O[t][r] / lrow[r];
      AO[((size_t)b * SEQ + m) * D_MODEL + h * D_HEAD + d] = (__bf16)v;
    }
  }
}

// ---------------------------------------------------------------------------
// Output projection:  out(fp32) = AO(bf16) @ Wo^T(bf16) + bo
// ---------------------------------------------------------------------------
__global__ __launch_bounds__(256) void mha_outproj_kernel(
    const __bf16* __restrict__ A, const __bf16* __restrict__ W,
    const float* __restrict__ bvec, float* __restrict__ out) {
  const int lane  = threadIdx.x & 31;
  const int wave  = threadIdx.x >> 5;
  const int half  = lane >> 4;
  const int nlane = lane & 15;
  const int m0 = blockIdx.x * 128 + wave * 16;
  const int n0 = blockIdx.y * 128;

  f32x8 C[8];
#pragma unroll
  for (int t = 0; t < 8; ++t)
#pragma unroll
    for (int r = 0; r < 8; ++r) C[t][r] = 0.f;

  for (int kk = 0; kk < D_MODEL; kk += 32) {
    bf16x16 a = load_a_bf16(A + (size_t)m0 * D_MODEL + kk, D_MODEL);
#pragma unroll
    for (int t = 0; t < 8; ++t) {
      bf16x16 bt = load_b_bf16(W + (size_t)(n0 + t * 16) * D_MODEL + kk, D_MODEL);
      C[t] = wmma_bf16(a, bt, C[t]);
    }
  }

#pragma unroll
  for (int t = 0; t < 8; ++t) {
    const int n = n0 + t * 16 + nlane;
    const float bb = bvec[n];
#pragma unroll
    for (int r = 0; r < 8; ++r) {
      const int m = m0 + r + 8 * half;
      out[(size_t)m * D_MODEL + n] = C[t][r] + bb;
    }
  }
}

// ---------------------------------------------------------------------------
extern "C" void kernel_launch(void* const* d_in, const int* in_sizes, int n_in,
                              void* d_out, int out_size, void* d_ws, size_t ws_size,
                              hipStream_t stream) {
  const float* query = (const float*)d_in[0];
  const float* key_  = (const float*)d_in[1];
  const float* value = (const float*)d_in[2];
  const int*   mask  = (const int*)d_in[3];
  const float* bias  = (const float*)d_in[4];
  const float* Wq = (const float*)d_in[5];
  const float* bq = (const float*)d_in[6];
  const float* Wk = (const float*)d_in[7];
  const float* bk = (const float*)d_in[8];
  const float* Wv = (const float*)d_in[9];
  const float* bv = (const float*)d_in[10];
  const float* Wo = (const float*)d_in[11];
  const float* bo = (const float*)d_in[12];
  float* out = (float*)d_out;

  // workspace layout (bf16 buffers)
  char* ws = (char*)d_ws;
  const size_t ACT = (size_t)M_TOTAL * D_MODEL * sizeof(__bf16);   // 8 MB
  const size_t WGT = (size_t)D_MODEL * D_MODEL * sizeof(__bf16);   // 2 MB
  __bf16* Qb  = (__bf16*)(ws + 0 * ACT);        // [B,H,S,dh]
  __bf16* Kb  = (__bf16*)(ws + 1 * ACT);        // [B,H,S,dh]
  __bf16* Vtb = (__bf16*)(ws + 2 * ACT);        // [B,H,dh,S]
  __bf16* AO  = (__bf16*)(ws + 3 * ACT);        // [B,S,H*dh]
  __bf16* Xq  = (__bf16*)(ws + 4 * ACT);
  __bf16* Xk  = (__bf16*)(ws + 5 * ACT);
  __bf16* Xv  = (__bf16*)(ws + 6 * ACT);
  __bf16* Wqb = (__bf16*)(ws + 7 * ACT + 0 * WGT);
  __bf16* Wkb = (__bf16*)(ws + 7 * ACT + 1 * WGT);
  __bf16* Wvb = (__bf16*)(ws + 7 * ACT + 2 * WGT);
  __bf16* Wob = (__bf16*)(ws + 7 * ACT + 3 * WGT);

  // fp32 -> bf16 conversions (activations: 4M elems; weights: 1M elems)
  const int nAct8 = (M_TOTAL * D_MODEL) / 8, nWgt8 = (D_MODEL * D_MODEL) / 8;
  mha_cvt_bf16_kernel<<<nAct8 / 256, 256, 0, stream>>>(query, Xq, nAct8);
  mha_cvt_bf16_kernel<<<nAct8 / 256, 256, 0, stream>>>(key_, Xk, nAct8);
  mha_cvt_bf16_kernel<<<nAct8 / 256, 256, 0, stream>>>(value, Xv, nAct8);
  mha_cvt_bf16_kernel<<<nWgt8 / 256, 256, 0, stream>>>(Wq, Wqb, nWgt8);
  mha_cvt_bf16_kernel<<<nWgt8 / 256, 256, 0, stream>>>(Wk, Wkb, nWgt8);
  mha_cvt_bf16_kernel<<<nWgt8 / 256, 256, 0, stream>>>(Wv, Wvb, nWgt8);
  mha_cvt_bf16_kernel<<<nWgt8 / 256, 256, 0, stream>>>(Wo, Wob, nWgt8);

  dim3 gproj(M_TOTAL / 128, D_MODEL / 128);   // (32, 8)
  mha_proj_kernel<<<gproj, 256, 0, stream>>>(Xq, Wqb, bq, Qb, 0);
  mha_proj_kernel<<<gproj, 256, 0, stream>>>(Xk, Wkb, bk, Kb, 0);
  mha_proj_kernel<<<gproj, 256, 0, stream>>>(Xv, Wvb, bv, Vtb, 1);

  dim3 gattn(BATCH * N_HEADS, SEQ / 64);      // (32, 32)
  mha_flash_attn_kernel<<<gattn, 128, 0, stream>>>(Qb, Kb, Vtb, bias, mask, AO);

  mha_outproj_kernel<<<gproj, 256, 0, stream>>>(AO, Wob, bo, out);
}
